// WindowAttention_87917980549610
// MI455X (gfx1250) — matrix operand order repaired and checked by reading, not compiled, not run on hardware
//
#include <hip/hip_runtime.h>

// ---------------- problem constants ----------------
#define H_    16
#define HD_   32
#define NTOK  49
#define DIMM  512
#define QKV3  1536
#define KCONV 15
#define SCALE 0.17677669529663687f   // 32^-0.5

typedef _Float16 v16h __attribute__((ext_vector_type(16)));
typedef float    v8f  __attribute__((ext_vector_type(8)));
typedef unsigned int u32x4 __attribute__((ext_vector_type(4)));
typedef int      i32x4 __attribute__((ext_vector_type(4)));
typedef int      i32x8 __attribute__((ext_vector_type(8)));

// ---------------- LDS layout (bytes) ----------------
// phase1: stage (fp32 x window, 100352B) lives in qkvs region;
//         xs[64][512] f16 (65536B, zero-padded rows 49..63) lives in the sc region.
#define OFF_QKVS 0                 // 49*1536*2          = 150528
#define OFF_SC   150528            // 8 * 49*49*4        =  76832  (xs overlaps here, phases 1/2)
#define OFF_PB   227360            // 8 * 49*49*2        =  38416
#define OFF_CTX  265776            // 49*512*2           =  50176
#define OFF_PEW  315952            // 16*15*4            =    960
#define SMEM_SZ  316912

// TDM availability / arity selection
#if defined(__gfx1250__) && __has_builtin(__builtin_amdgcn_tensor_load_to_lds) && __has_builtin(__builtin_amdgcn_s_wait_tensorcnt)
#define CDNA5_USE_TDM 1
#endif
#if __has_include(<hip/amd_detail/amd_gfx1250_TDM.h>)
#define CDNA5_TDM6 1
#endif

struct alignas(16) H8v { _Float16 h[8]; };

static __device__ __forceinline__ v16h zfrag() {
  v16h a;
#pragma unroll
  for (int i = 0; i < 16; ++i) a[i] = (_Float16)0.f;
  return a;
}

// A-fragment (16xK tile, row-major source). p = &src[row][k0 + hi*16].
// element e -> K = k0 + 16*hi + e
static __device__ __forceinline__ v16h ld_a(const _Float16* p) {
  H8v c0 = *(const H8v*)p;
  H8v c1 = *(const H8v*)(p + 8);
  v16h a;
#pragma unroll
  for (int i = 0; i < 8; ++i) { a[i] = c0.h[i]; a[i + 8] = c1.h[i]; }
  return a;
}

// B-fragment for NT gemm (source stored [N][K] row-major). p = &src[ncol][k0 + hi*16].
// element e -> K = k0 + 16*hi + e
static __device__ __forceinline__ v16h ld_bt(const _Float16* p) {
  H8v c0 = *(const H8v*)p;
  H8v c1 = *(const H8v*)(p + 8);
  v16h b;
#pragma unroll
  for (int i = 0; i < 8; ++i) { b[i] = c0.h[i]; b[i + 8] = c1.h[i]; }
  return b;
}

static __device__ __forceinline__ v8f wmma_f16(v16h a, v16h b, v8f c) {
  return __builtin_amdgcn_wmma_f32_16x16x32_f16(false, a, false, b, (short)0, c, false, false);
}

static __device__ __forceinline__ v8f zacc() {
  v8f c;
#pragma unroll
  for (int i = 0; i < 8; ++i) c[i] = 0.f;
  return c;
}

// ---------------- weight fp32 -> fp16 convert ----------------
extern "C" __global__ void cvt_weights(const float* __restrict__ qkvw,
                                       const float* __restrict__ projw,
                                       _Float16* __restrict__ wh,
                                       _Float16* __restrict__ pwh) {
  int i = blockIdx.x * 256 + threadIdx.x;
  if (i < QKV3 * DIMM) wh[i] = (_Float16)qkvw[i];
  if (i < DIMM * DIMM) pwh[i] = (_Float16)projw[i];
}

// ---------------- fused window attention (one block per window) ----------------
extern "C" __global__ __launch_bounds__(256)
void winattn_fused(const float* __restrict__ x,      // (2048,49,512)
                   const float* __restrict__ mask,   // (64,49,49)
                   const float* __restrict__ qkv_b,  // (1536,)
                   const float* __restrict__ proj_b, // (512,)
                   const float* __restrict__ rpb,    // (169,16)
                   const float* __restrict__ pe_w,   // (16,1,15,1)
                   const float* __restrict__ pe_b,   // (16,)
                   const int*   __restrict__ rpi,    // (49,49)
                   const _Float16* __restrict__ wh,  // (1536,512) f16
                   const _Float16* __restrict__ pwh, // (512,512)  f16
                   float* __restrict__ out)          // (2048,49,512)
{
  __shared__ alignas(16) unsigned char smem[SMEM_SZ];
  _Float16* qkvs = (_Float16*)(smem + OFF_QKVS);                  // [49][1536]
  _Float16* xs   = (_Float16*)(smem + OFF_SC);                    // [64][512], rows 49..63 zero
  _Float16* ctx  = (_Float16*)(smem + OFF_CTX);                   // [49][512]
  float*    pew  = (float*)(smem + OFF_PEW);                      // [16][15]

  const int bwin = blockIdx.x;
  const int tid  = threadIdx.x;
  const int wv   = tid >> 5;
  const int lane = tid & 31;
  const int l15  = lane & 15;
  const int hi   = (lane >> 4) & 1;

  const float* xw = x + (size_t)bwin * NTOK * DIMM;

  // ---------------- phase 1: x window -> LDS (TDM) -> f16, zero-pad rows 49..63 ----------------
#if defined(CDNA5_USE_TDM)
  if (tid == 0) {
    unsigned long long ga = (unsigned long long)(const void*)xw;
    u32x4 g0 = { 1u,                                   // count=1 (valid user descriptor)
                 (unsigned)OFF_QKVS,                   // lds_addr (staging at blob base)
                 (unsigned)(ga & 0xFFFFFFFFu),         // global_addr[31:0]
                 (unsigned)((ga >> 32) & 0x1FFFFFFu) | (2u << 30) }; // addr[56:32] | type=2
    i32x8 g1 = { (2 << 16),          // data_size=2 (4B elements)
                 (512 << 16),        // tensor_dim0 = 512 (bits 79:48)
                 (NTOK << 16),       // tensor_dim1 = 49  (bits 111:80)
                 (512 << 16),        // tile_dim0   = 512 (bits 127:112)
                 NTOK,               // tile_dim1   = 49  (bits 143:128)
                 512,                // tensor_dim0_stride = 512 (bits 207:160)
                 0, 0 };
    i32x4 gz4 = { 0, 0, 0, 0 };
#if defined(CDNA5_TDM6)
    i32x8 gz8 = { 0, 0, 0, 0, 0, 0, 0, 0 };
    __builtin_amdgcn_tensor_load_to_lds(g0, g1, gz4, gz4, gz8, 0);
#else
    __builtin_amdgcn_tensor_load_to_lds(g0, g1, gz4, gz4, 0);
#endif
    __builtin_amdgcn_s_wait_tensorcnt(0);
  }
  if (tid < H_ * KCONV) pew[tid] = pe_w[tid];
  __syncthreads();
  {
    const float* stage = (const float*)(smem + OFF_QKVS);
    for (int i = tid; i < NTOK * DIMM; i += 256) xs[i] = (_Float16)stage[i];
    for (int i = tid; i < (64 - NTOK) * DIMM; i += 256) xs[NTOK * DIMM + i] = (_Float16)0.f;
  }
  __syncthreads();
#else
  for (int i = tid; i < NTOK * DIMM; i += 256) xs[i] = (_Float16)xw[i];
  for (int i = tid; i < (64 - NTOK) * DIMM; i += 256) xs[NTOK * DIMM + i] = (_Float16)0.f;
  if (tid < H_ * KCONV) pew[tid] = pe_w[tid];
  __syncthreads();
#endif

  // ---------------- phase 2: qkv = xs @ wh^T + b   (49x1536, K=512) ----------------
  // wave wv owns columns [wv*192, wv*192+192); xs is zero-padded so A loads are unconditional
  for (int jt = 0; jt < 12; ++jt) {
    const int col  = wv * 192 + jt * 16 + l15;
    const float bc = qkv_b[col];
    if (jt < 11) __builtin_prefetch(wh + (size_t)(col + 16) * DIMM, 0, 0);
    v8f acc[4];
#pragma unroll
    for (int it = 0; it < 4; ++it) acc[it] = zacc();
#pragma unroll 1
    for (int kk = 0; kk < 16; ++kk) {
      const int k0 = kk * 32;
      const v16h bF = ld_bt(wh + (size_t)col * DIMM + k0 + hi * 16);
#pragma unroll
      for (int it = 0; it < 4; ++it) {
        const v16h aF = ld_a(xs + (it * 16 + l15) * DIMM + k0 + hi * 16);
        acc[it] = wmma_f16(aF, bF, acc[it]);
      }
    }
#pragma unroll
    for (int it = 0; it < 4; ++it)
#pragma unroll
      for (int r = 0; r < 8; ++r) {
        const int row = it * 16 + r + hi * 8;
        if (row < NTOK) qkvs[row * QKV3 + col] = (_Float16)(acc[it][r] + bc);
      }
  }
  __syncthreads();

  // ---------------- phase 3/4: per-head attention (wave-private) ----------------
  float* scw  = (float*)(smem + OFF_SC) + wv * NTOK * NTOK;       // [49][49] f32 scores
  _Float16* pb = (_Float16*)(smem + OFF_PB) + wv * NTOK * NTOK;   // [49][49] f16 probs
  const float* msk = mask + (size_t)(bwin & 63) * NTOK * NTOK;

  for (int hp = 0; hp < 2; ++hp) {
    const int h  = wv + hp * 8;
    const int qo = h * HD_;
    const int ko = DIMM + h * HD_;
    const int vo = 2 * DIMM + h * HD_;

    // scores = (q*scale) @ k^T  : K = HD = 32, single wmma step per tile
    for (int it = 0; it < 4; ++it) {
      const int row = it * 16 + l15;
      const v16h aF = (row < NTOK) ? ld_a(qkvs + row * QKV3 + qo + hi * 16) : zfrag();
      for (int jt = 0; jt < 4; ++jt) {
        const int m  = jt * 16 + l15;
        const v16h bF = (m < NTOK) ? ld_bt(qkvs + m * QKV3 + ko + hi * 16) : zfrag();
        v8f c = zacc();
        c = wmma_f16(aF, bF, c);
        if (m < NTOK) {
#pragma unroll
          for (int r = 0; r < 8; ++r) {
            const int rw = it * 16 + r + hi * 8;
            if (rw < NTOK) scw[rw * NTOK + m] = c[r] * SCALE;
          }
        }
      }
    }

    // depthwise conv (15 taps along query axis) + rel-pos bias + mask + softmax
    const float pebias = pe_b[h];
    float wpe[KCONV];
#pragma unroll
    for (int t = 0; t < KCONV; ++t) wpe[t] = pew[h * KCONV + t];

    for (int n = 0; n < NTOK; ++n) {
      const int m0 = lane;
      const int m1 = lane + 32;
      const bool v1 = (m1 < NTOK);
      float pe0 = 0.f, pe1 = 0.f;
#pragma unroll
      for (int t = 0; t < KCONV; ++t) {
        const int nn = n + t - 7;
        if (nn >= 0 && nn < NTOK) {
          pe0 += wpe[t] * scw[nn * NTOK + m0];
          if (v1) pe1 += wpe[t] * scw[nn * NTOK + m1];
        }
      }
      float a0 = scw[n * NTOK + m0] + pe0 + pebias
               + rpb[rpi[n * NTOK + m0] * H_ + h] + msk[n * NTOK + m0];
      float a1 = v1 ? (scw[n * NTOK + m1] + pe1 + pebias
               + rpb[rpi[n * NTOK + m1] * H_ + h] + msk[n * NTOK + m1])
                    : -3.0e38f;
      float mx = fmaxf(a0, a1);
#pragma unroll
      for (int o = 16; o; o >>= 1) mx = fmaxf(mx, __shfl_xor(mx, o, 32));
      const float e0 = __expf(a0 - mx);
      const float e1 = v1 ? __expf(a1 - mx) : 0.f;
      float s = e0 + e1;
#pragma unroll
      for (int o = 16; o; o >>= 1) s += __shfl_xor(s, o, 32);
      const float inv = 1.f / s;
      pb[n * NTOK + m0] = (_Float16)(e0 * inv);
      if (v1) pb[n * NTOK + m1] = (_Float16)(e1 * inv);
    }

    // ctx(49x32) = P(49x49) @ V(49x32) : K padded to 64, zeros beyond 49
    for (int it = 0; it < 4; ++it) {
      for (int jd = 0; jd < 2; ++jd) {
        v8f c = zacc();
        for (int kk = 0; kk < 2; ++kk) {
          const int row = it * 16 + l15;
          const bool rv = (row < NTOK);
          v16h aF, bF;
#pragma unroll
          for (int e = 0; e < 16; ++e) {
            const int kA = kk * 32 + hi * 16 + e;
            aF[e] = (rv && kA < NTOK) ? pb[row * NTOK + kA] : (_Float16)0.f;
            const int kB = kk * 32 + hi * 16 + e;
            bF[e] = (kB < NTOK) ? qkvs[kB * QKV3 + vo + jd * 16 + l15] : (_Float16)0.f;
          }
          c = wmma_f16(aF, bF, c);
        }
        const int col = h * HD_ + jd * 16 + l15;
#pragma unroll
        for (int r = 0; r < 8; ++r) {
          const int rw = it * 16 + r + hi * 8;
          if (rw < NTOK) ctx[rw * DIMM + col] = (_Float16)c[r];
        }
      }
    }
  }
  __syncthreads();

  // ---------------- phase 5: out = ctx @ proj_w^T + b   (49x512, K=512) ----------------
  for (int ct = 0; ct < 4; ++ct) {
    const int col  = (wv * 4 + ct) * 16 + l15;
    const float pc = proj_b[col];
    if (ct < 3) __builtin_prefetch(pwh + (size_t)(col + 16) * DIMM, 0, 0);
    v8f acc[4];
#pragma unroll
    for (int it = 0; it < 4; ++it) acc[it] = zacc();
#pragma unroll 1
    for (int kk = 0; kk < 16; ++kk) {
      const int k0 = kk * 32;
      const v16h bF = ld_bt(pwh + (size_t)col * DIMM + k0 + hi * 16);
#pragma unroll
      for (int it = 0; it < 4; ++it) {
        const int row = it * 16 + l15;
        const v16h aF = (row < NTOK) ? ld_a(ctx + row * DIMM + k0 + hi * 16) : zfrag();
        acc[it] = wmma_f16(aF, bF, acc[it]);
      }
    }
#pragma unroll
    for (int it = 0; it < 4; ++it)
#pragma unroll
      for (int r = 0; r < 8; ++r) {
        const int rw = it * 16 + r + hi * 8;
        if (rw < NTOK)
          out[((size_t)bwin * NTOK + rw) * DIMM + col] = acc[it][r] + pc;
      }
  }
}

// ---------------- host entry ----------------
extern "C" void kernel_launch(void* const* d_in, const int* in_sizes, int n_in,
                              void* d_out, int out_size, void* d_ws, size_t ws_size,
                              hipStream_t stream) {
  (void)in_sizes; (void)n_in; (void)out_size; (void)ws_size;
  const float* x      = (const float*)d_in[0];
  const float* mask   = (const float*)d_in[1];
  const float* qkv_w  = (const float*)d_in[2];
  const float* qkv_b  = (const float*)d_in[3];
  const float* proj_w = (const float*)d_in[4];
  const float* proj_b = (const float*)d_in[5];
  const float* rpb    = (const float*)d_in[6];
  const float* pe_w   = (const float*)d_in[7];
  const float* pe_b   = (const float*)d_in[8];
  const int*   rpi    = (const int*)d_in[9];

  _Float16* wh  = (_Float16*)d_ws;                                        // 1536*512 f16
  _Float16* pwh = (_Float16*)((char*)d_ws + (size_t)QKV3 * DIMM * 2);     // 512*512 f16

  cvt_weights<<<(QKV3 * DIMM + 255) / 256, 256, 0, stream>>>(qkv_w, proj_w, wh, pwh);
  winattn_fused<<<2048, 256, 0, stream>>>(x, mask, qkv_b, proj_b, rpb,
                                          pe_w, pe_b, rpi, wh, pwh, (float*)d_out);
}
